// CrossAttention_60962765799888
// MI455X (gfx1250) — matrix operand order repaired and checked by reading, not compiled, not run on hardware
//
#include <hip/hip_runtime.h>
#include <hip/hip_bf16.h>
#include <cstdint>

// ---------------------------------------------------------------------------
// Types matching the gfx1250 WMMA builtin signatures
// ---------------------------------------------------------------------------
typedef __bf16 bf16_t;
typedef __attribute__((ext_vector_type(8)))  __bf16 v8bf;
typedef __attribute__((ext_vector_type(16))) __bf16 v16bf;
typedef __attribute__((ext_vector_type(8)))  float  v8f;
typedef __attribute__((ext_vector_type(4)))  unsigned int v4u;
typedef __attribute__((ext_vector_type(4)))  int v4i;

// D(16x16,f32) = A(16x32,bf16) * B(32x16,bf16) + C
#define WMMA_BF16(a, b, c) \
  __builtin_amdgcn_wmma_f32_16x16x32_bf16(false, (a), false, (b), (short)0, (c), false, false)

// Problem constants (from reference)
#define BATCH   4
#define SQ      4096
#define SKV     1024
#define DE      1024     // d_embed
#define DC      768      // d_cross
#define NH      16
#define DH      64       // head dim

// ---------------------------------------------------------------------------
// gfx1250 async global->LDS staging (ASYNCcnt path).
// ---------------------------------------------------------------------------
#ifndef __has_builtin
#define __has_builtin(x) 0
#endif
#if __has_builtin(__builtin_amdgcn_global_load_async_to_lds_b128) && \
    __has_builtin(__builtin_amdgcn_s_wait_asynccnt)
#define USE_ASYNC_LDS 1
#else
#define USE_ASYNC_LDS 0
#endif

#if USE_ASYNC_LDS
typedef __attribute__((address_space(1))) v4i gv4i_t;
typedef __attribute__((address_space(3))) v4i lv4i_t;
#endif

__device__ __forceinline__ void async_wait_all() {
#if USE_ASYNC_LDS
  __builtin_amdgcn_s_wait_asynccnt(0);
#endif
}

// copy 16 bf16 (32B) global -> LDS, per-lane
__device__ __forceinline__ void copy16_bf16(const bf16_t* __restrict__ g, bf16_t* l) {
#if USE_ASYNC_LDS
  gv4i_t* gp0 = (gv4i_t*)(uintptr_t)(g);
  gv4i_t* gp1 = (gv4i_t*)(uintptr_t)(g + 8);
  lv4i_t* lp0 = (lv4i_t*)(uintptr_t)(l);
  lv4i_t* lp1 = (lv4i_t*)(uintptr_t)(l + 8);
  __builtin_amdgcn_global_load_async_to_lds_b128(gp0, lp0, 0, 0);
  __builtin_amdgcn_global_load_async_to_lds_b128(gp1, lp1, 0, 0);
#else
  *(v4u*)l       = *(const v4u*)g;
  *(v4u*)(l + 8) = *(const v4u*)(g + 8);
#endif
}

// ---------------------------------------------------------------------------
// f32 -> bf16 weight conversion
// ---------------------------------------------------------------------------
__global__ void cvt_f32_bf16_kernel(const float* __restrict__ src,
                                    bf16_t* __restrict__ dst, int n) {
  int i = blockIdx.x * blockDim.x + threadIdx.x;
  int stride = gridDim.x * blockDim.x;
  for (; i < n; i += stride) dst[i] = (bf16_t)src[i];
}

// ---------------------------------------------------------------------------
// LDS staging helpers
// ---------------------------------------------------------------------------
__device__ __forceinline__ void load_a8(const float* __restrict__ p, bf16_t* dst) {
  float4 f0 = *(const float4*)p;
  float4 f1 = *(const float4*)(p + 4);
  dst[0] = (bf16_t)f0.x; dst[1] = (bf16_t)f0.y; dst[2] = (bf16_t)f0.z; dst[3] = (bf16_t)f0.w;
  dst[4] = (bf16_t)f1.x; dst[5] = (bf16_t)f1.y; dst[6] = (bf16_t)f1.z; dst[7] = (bf16_t)f1.w;
}

// stage 16 A elements: f32 path converts, bf16 path async-copies
__device__ __forceinline__ void stage_a16(const float* __restrict__ g, bf16_t* l) {
  load_a8(g, l);
  load_a8(g + 8, l + 8);
}
__device__ __forceinline__ void stage_a16(const bf16_t* __restrict__ g, bf16_t* l) {
  copy16_bf16(g, l);
}

// Pack two contiguous 8-half chunks into a WMMA fragment
__device__ __forceinline__ v16bf pack_frag(const bf16_t* lo_p, const bf16_t* hi_p) {
  v8bf lo = *(const v8bf*)lo_p;
  v8bf hi = *(const v8bf*)hi_p;
  v16bf f;
#pragma unroll
  for (int i = 0; i < 8; ++i) { f[i] = lo[i]; f[i + 8] = hi[i]; }
  return f;
}

// ---------------------------------------------------------------------------
// GEMM with bias: C[M,N] = A[M,K] * B[K,N] + bias[N]
// Block: 256 threads (8 waves). Block tile: 128(M) x 128(N), BK = 32,
// double-buffered LDS with async global->LDS staging.
// ---------------------------------------------------------------------------
template <typename AT, typename OT>
__global__ __launch_bounds__(256)
void gemm_bias_wmma_kernel(const AT* __restrict__ A, const bf16_t* __restrict__ B,
                           const float* __restrict__ bias, OT* __restrict__ C,
                           int M, int N, int K) {
  __shared__ __align__(32) bf16_t As[2][128][32];   // 16 KB
  __shared__ __align__(32) bf16_t Bs[2][32][128];   // 16 KB

  const int tid  = threadIdx.x;
  const int lane = tid & 31;
  const int wave = tid >> 5;
  const int wm   = wave & 3;   // 4 waves along M (32 rows each)
  const int wn   = wave >> 2;  // 2 waves along N (64 cols each)
  const int m0   = blockIdx.x * 128;
  const int n0   = blockIdx.y * 128;

  v8f acc[2][4] = {};          // [m-subtile][n-subtile]

  const int arow = tid >> 1, acol = (tid & 1) * 16;
  const int brow = tid >> 3, bcol = (tid & 7) * 16;

  const int a_klo  = (lane < 16) ? 0 : 8;
  const int a_row0 = wm * 32 + (lane & 15);

  // --- prologue: stage first K tile ---
  stage_a16(A + (size_t)(m0 + arow) * K + acol, &As[0][arow][acol]);
  copy16_bf16(B + (size_t)brow * N + n0 + bcol, &Bs[0][brow][bcol]);
  async_wait_all();
  __syncthreads();

  int cur = 0;
  for (int k0 = 0; k0 < K; k0 += 32) {
    if (k0 + 32 < K) {
      stage_a16(A + (size_t)(m0 + arow) * K + (k0 + 32) + acol, &As[cur ^ 1][arow][acol]);
      copy16_bf16(B + (size_t)(k0 + 32 + brow) * N + n0 + bcol, &Bs[cur ^ 1][brow][bcol]);
    }

    v16bf af[2];
#pragma unroll
    for (int mt = 0; mt < 2; ++mt) {
      const bf16_t* ar = &As[cur][a_row0 + mt * 16][a_klo];
      af[mt] = pack_frag(ar, ar + 16);
    }

#pragma unroll
    for (int nt = 0; nt < 4; ++nt) {
      v16bf bfr;
#pragma unroll
      for (int i = 0; i < 8; ++i) {
        bfr[i]     = Bs[cur][a_klo + i][wn * 64 + nt * 16 + (lane & 15)];
        bfr[i + 8] = Bs[cur][16 + a_klo + i][wn * 64 + nt * 16 + (lane & 15)];
      }
#pragma unroll
      for (int mt = 0; mt < 2; ++mt)
        acc[mt][nt] = WMMA_BF16(af[mt], bfr, acc[mt][nt]);
    }

    async_wait_all();
    __syncthreads();
    cur ^= 1;
  }

  const int col = n0 + wn * 64 + (lane & 15);
#pragma unroll
  for (int mt = 0; mt < 2; ++mt) {
    const int row_base = m0 + wm * 32 + mt * 16 + ((lane < 16) ? 0 : 8);
#pragma unroll
    for (int nt = 0; nt < 4; ++nt) {
      float bv = bias[col + nt * 16];
#pragma unroll
      for (int r = 0; r < 8; ++r)
        C[(size_t)(row_base + r) * N + col + nt * 16] = (OT)(acc[mt][nt][r] + bv);
    }
  }
}

// ---------------------------------------------------------------------------
// Flash attention via S^T = K * Q^T.
// One block = (batch b, head h, 64 q rows), 128 threads = 4 waves; wave owns
// 16 q values. A-frags (K rows) and V B-frags come straight from global
// (contiguous, cache-resident); Q^T B-frags gathered once per block; the
// kv loop has NO barriers and NO shared staging.
// In S^T C-layout: M = kv (VGPR rows, halves split kv +8), N = q (lane).
// Softmax over kv for fixed q = 31 in-lane ops + one shfl_xor(16).
// ---------------------------------------------------------------------------
__global__ __launch_bounds__(128)
void flash_attn_wmma_kernel(const bf16_t* __restrict__ Q,
                            const bf16_t* __restrict__ Km,
                            const bf16_t* __restrict__ Vm,
                            bf16_t* __restrict__ Oa) {
  const int qt   = blockIdx.x;
  const int h    = blockIdx.y;
  const int b    = blockIdx.z;
  const int tid  = threadIdx.x;
  const int lane = tid & 31;
  const int wave = tid >> 5;
  const int q0   = qt * 64;
  const int hcol = h * DH;
  const int l16  = lane & 15;
  const int hi8  = (lane < 16) ? 0 : 8;
  const int klo  = hi8;                 // A-frag K-chunk offset per ISA layout

  __shared__ __align__(32) bf16_t Pl[4][16][64];   // per-wave P [q][kv], 8 KB
  __shared__ float scw[4][16];                     // per-q rescale broadcast
  __shared__ float lw[4][16];                      // per-q 1/l broadcast

  const float scale = 0.125f;           // 1/sqrt(64)

  // --- B-frags of Q^T (kv-invariant, built once): lane%16 = q col, elements = d (k) ---
  v16bf bq[2];
#pragma unroll
  for (int ks = 0; ks < 2; ++ks) {
    const bf16_t* qp = Q + ((size_t)b * SQ + q0 + wave * 16 + l16) * DE + hcol + ks * 32 + klo;
    bq[ks] = pack_frag(qp, qp + 16);
  }

  v8f o[4] = {};                        // O: 16(q rows) x 64(d), C layout
  float mq = -1e30f, lq = 0.0f;         // per-lane state for q = wave*16 + l16

  for (int kt = 0; kt < SKV / 64; ++kt) {
    const int kv0 = kt * 64;

    // --- S^T: 4 kv-subtiles x (this wave's 16 q); A = K rows from global ---
    v8f s[4];
#pragma unroll
    for (int mt = 0; mt < 4; ++mt) {
      v8f a = {};
#pragma unroll
      for (int ks = 0; ks < 2; ++ks) {
        const bf16_t* kp = Km + ((size_t)b * SKV + kv0 + mt * 16 + l16) * DE
                              + hcol + ks * 32 + klo;
        v16bf ak = pack_frag(kp, kp + 16);
        a = WMMA_BF16(ak, bq[ks], a);
      }
      s[mt] = a;
    }

    // --- online softmax for q = lane's column ---
    float tmax = s[0][0];
#pragma unroll
    for (int mt = 0; mt < 4; ++mt)
#pragma unroll
      for (int r = 0; r < 8; ++r) tmax = fmaxf(tmax, s[mt][r]);
    tmax = fmaxf(tmax, __shfl_xor(tmax, 16, 32));
    tmax *= scale;

    const float mnew = fmaxf(mq, tmax);
    const float sc   = __expf(mq - mnew);
    float tsum = 0.0f;
#pragma unroll
    for (int mt = 0; mt < 4; ++mt)
#pragma unroll
      for (int r = 0; r < 8; ++r) {
        float p = __expf(s[mt][r] * scale - mnew);
        s[mt][r] = p;
        tsum += p;
      }
    tsum += __shfl_xor(tsum, 16, 32);
    lq = lq * sc + tsum;
    mq = mnew;

    // --- broadcast per-q rescale to O's row layout; pack P (8 contiguous kv
    //     per store -> one b128 per subtile) into per-wave LDS [q][kv] ---
    if (lane < 16) scw[wave][l16] = sc;
#pragma unroll
    for (int mt = 0; mt < 4; ++mt) {
      v8bf t;
#pragma unroll
      for (int r = 0; r < 8; ++r) t[r] = (bf16_t)s[mt][r];
      *(v8bf*)&Pl[wave][l16][mt * 16 + hi8] = t;
    }

#pragma unroll
    for (int r = 0; r < 8; ++r) {
      float srow = scw[wave][r + hi8];
#pragma unroll
      for (int dt = 0; dt < 4; ++dt) o[dt][r] *= srow;
    }

    // --- O += P * V : P A-frags from LDS, V B-frags straight from global ---
    v16bf ap[2];
#pragma unroll
    for (int ks = 0; ks < 2; ++ks) {
      const bf16_t* pp = &Pl[wave][l16][ks * 32 + klo];
      ap[ks] = pack_frag(pp, pp + 16);
    }
#pragma unroll
    for (int dt = 0; dt < 4; ++dt)
#pragma unroll
      for (int ks = 0; ks < 2; ++ks) {
        const bf16_t* vp = Vm + ((size_t)b * SKV + kv0 + ks * 32 + klo) * DE
                              + hcol + dt * 16 + l16;
        v16bf bv;
#pragma unroll
        for (int i = 0; i < 8; ++i) {
          bv[i]     = vp[(size_t)i * DE];
          bv[i + 8] = vp[(size_t)(16 + i) * DE];
        }
        o[dt] = WMMA_BF16(ap[ks], bv, o[dt]);
      }
  }

  // --- normalize (broadcast 1/l to row layout) and store bf16 output ---
  if (lane < 16) lw[wave][l16] = 1.0f / lq;
  const int row_base = q0 + wave * 16 + hi8;
#pragma unroll
  for (int r = 0; r < 8; ++r) {
    float inv = lw[wave][r + hi8];
#pragma unroll
    for (int dt = 0; dt < 4; ++dt)
      Oa[((size_t)b * SQ + row_base + r) * DE + hcol + dt * 16 + l16] =
          (bf16_t)(o[dt][r] * inv);
  }
}

// ---------------------------------------------------------------------------
// Launcher
// ---------------------------------------------------------------------------
extern "C" void kernel_launch(void* const* d_in, const int* in_sizes, int n_in,
                              void* d_out, int out_size, void* d_ws, size_t ws_size,
                              hipStream_t stream) {
  (void)in_sizes; (void)n_in; (void)out_size; (void)ws_size;

  const float* x  = (const float*)d_in[0];  // [4,4096,1024]
  const float* y  = (const float*)d_in[1];  // [4,1024,768]
  const float* Wq = (const float*)d_in[2];
  const float* bq = (const float*)d_in[3];
  const float* Wk = (const float*)d_in[4];
  const float* bk = (const float*)d_in[5];
  const float* Wv = (const float*)d_in[6];
  const float* bv = (const float*)d_in[7];
  const float* Wo = (const float*)d_in[8];
  const float* bo = (const float*)d_in[9];
  float* out = (float*)d_out;               // [4,4096,1024]

  // workspace carve-out (all bf16), 256B aligned
  size_t off = 0;
  auto carve = [&](size_t bytes) {
    void* p = (char*)d_ws + off;
    off += (bytes + 255) & ~(size_t)255;
    return p;
  };
  bf16_t* Wqb = (bf16_t*)carve((size_t)DE * DE * 2);
  bf16_t* Wkb = (bf16_t*)carve((size_t)DC * DE * 2);
  bf16_t* Wvb = (bf16_t*)carve((size_t)DC * DE * 2);
  bf16_t* Wob = (bf16_t*)carve((size_t)DE * DE * 2);
  bf16_t* Qb  = (bf16_t*)carve((size_t)BATCH * SQ  * DE * 2);
  bf16_t* Kb  = (bf16_t*)carve((size_t)BATCH * SKV * DE * 2);
  bf16_t* Vb  = (bf16_t*)carve((size_t)BATCH * SKV * DE * 2);
  bf16_t* Ab  = (bf16_t*)carve((size_t)BATCH * SQ  * DE * 2);

  // 1) weights -> bf16
  cvt_f32_bf16_kernel<<<1024, 256, 0, stream>>>(Wq, Wqb, DE * DE);
  cvt_f32_bf16_kernel<<<1024, 256, 0, stream>>>(Wk, Wkb, DC * DE);
  cvt_f32_bf16_kernel<<<1024, 256, 0, stream>>>(Wv, Wvb, DC * DE);
  cvt_f32_bf16_kernel<<<1024, 256, 0, stream>>>(Wo, Wob, DE * DE);

  // 2) projections (A converted f32->bf16 during LDS staging)
  {
    dim3 gq(BATCH * SQ / 128, DE / 128);
    gemm_bias_wmma_kernel<float, bf16_t><<<gq, 256, 0, stream>>>(x, Wqb, bq, Qb,
                                                                 BATCH * SQ, DE, DE);
    dim3 gk(BATCH * SKV / 128, DE / 128);
    gemm_bias_wmma_kernel<float, bf16_t><<<gk, 256, 0, stream>>>(y, Wkb, bk, Kb,
                                                                 BATCH * SKV, DE, DC);
    gemm_bias_wmma_kernel<float, bf16_t><<<gk, 256, 0, stream>>>(y, Wvb, bv, Vb,
                                                                 BATCH * SKV, DE, DC);
  }

  // 3) attention
  {
    dim3 ga(SQ / 64, NH, BATCH);
    flash_attn_wmma_kernel<<<ga, 128, 0, stream>>>(Qb, Kb, Vb, Ab);
  }

  // 4) output projection -> f32 d_out
  {
    dim3 go(BATCH * SQ / 128, DE / 128);
    gemm_bias_wmma_kernel<bf16_t, float><<<go, 256, 0, stream>>>(Ab, Wob, bo, out,
                                                                 BATCH * SQ, DE, DE);
  }
}
